// trajectory2seq_attn_87857851007520
// MI455X (gfx1250) — matrix operand order repaired and checked by reading, not compiled, not run on hardware
//
#include <hip/hip_runtime.h>

// ---------------- problem dims ----------------
constexpr int NB  = 64;    // batch
constexpr int NS  = 512;   // encoder sequence length
constexpr int NT  = 64;    // decoder steps
constexpr int NH  = 1024;  // hidden
constexpr int NV  = 1024;  // vocab
constexpr int NH3 = 3072;  // 3*H

constexpr int CHUNK = 128; // K-chunk staged in LDS (64 x 128 f16 = 16 KB per buffer)

typedef _Float16 f16;
typedef __attribute__((ext_vector_type(16))) _Float16 v16h;
typedef __attribute__((ext_vector_type(8)))  _Float16 v8h;
typedef __attribute__((ext_vector_type(8)))  float    v8f;

// Set to 0 if the gfx1250 async-to-LDS mnemonic fails to assemble.
#define USE_ASYNC_LDS 1

// cooperative copy of one 64 x CHUNK A-chunk into LDS (16B per lane per issue)
__device__ __forceinline__ void fill_chunk(f16* __restrict__ ldsbuf,
                                           const f16* __restrict__ A,
                                           int K, int kc, int tid)
{
    const int nvec = 64 * CHUNK / 8;            // 16-byte vectors in the chunk
#pragma unroll
    for (int v = tid; v < nvec; v += 256) {
        const int row = v / (CHUNK / 8);
        const int c8  = v - row * (CHUNK / 8);
        const f16* src = A + row * K + kc + c8 * 8;
        f16*       dst = ldsbuf + row * CHUNK + c8 * 8;
#if USE_ASYNC_LDS
        unsigned int       laddr = (unsigned int)(size_t)(void*)dst;        // LDS byte addr
        unsigned long long gaddr = (unsigned long long)(size_t)(const void*)src;
        asm volatile("global_load_async_to_lds_b128 %0, %1, off"
                     :: "v"(laddr), "v"(gaddr) : "memory");
#else
        *(v8h*)dst = *(const v8h*)src;
#endif
    }
}

__device__ __forceinline__ void wait_fill()
{
#if USE_ASYNC_LDS
    asm volatile("s_wait_asynccnt 0x0" ::: "memory");
#endif
}

// ---------------------------------------------------------------------------
// WMMA GEMM:  out[M=64, N] = A[64, K](f16) x W[N, K]^T(f16) + bias
// Block = 256 threads (8 waves) computes a 64x64 output tile.
// Wave computes 16(M) x 32(N): two v_wmma_f32_16x16x32_f16 accumulators
// sharing one A fragment. A is double-buffer staged in LDS via async copies.
// ---------------------------------------------------------------------------
__global__ __launch_bounds__(256)
void gemm_wmma_f16(const f16* __restrict__ A, const f16* __restrict__ W,
                   const float* __restrict__ bias,
                   float* __restrict__ out, int ldo,
                   f16* __restrict__ out16,
                   int N, int K)
{
    __shared__ f16 aLds[2][64 * CHUNK];         // 32 KB static LDS

    const int tid  = threadIdx.x;
    const int lane = tid & 31;
    const int wid  = tid >> 5;
    const int m0   = (wid & 3) << 4;            // 4 M-tiles cover M = 64
    const int n0   = blockIdx.x * 64 + (wid >> 2) * 32;
    const int half = lane >> 4;                 // lane group 0-15 / 16-31
    const int rc   = lane & 15;                 // A row / B column within tile

    const f16* wrow0 = W + (n0 + rc) * K + half * 16;
    const f16* wrow1 = W + (n0 + 16 + rc) * K + half * 16;

    v8f c0, c1;
    {
        const float b0 = bias ? bias[n0 + rc] : 0.0f;
        const float b1 = bias ? bias[n0 + 16 + rc] : 0.0f;
#pragma unroll
        for (int i = 0; i < 8; ++i) { c0[i] = b0; c1[i] = b1; }
    }

    const int nchunks = K / CHUNK;
    int buf = 0;
    fill_chunk(aLds[0], A, K, 0, tid);
    wait_fill();
    __syncthreads();

    for (int c = 0; c < nchunks; ++c) {
        if (c + 1 < nchunks)                     // prefetch next chunk into other buffer
            fill_chunk(aLds[buf ^ 1], A, K, (c + 1) * CHUNK, tid);

        const f16* aBase = aLds[buf] + (m0 + rc) * CHUNK + half * 16;
        const int  kg0   = c * CHUNK;
#pragma unroll
        for (int k0 = 0; k0 < CHUNK; k0 += 32) {
            // A lane data from LDS: K = k0 + half*16 + {0..15}
            v8h a_lo = *(const v8h*)(aBase + k0);
            v8h a_hi = *(const v8h*)(aBase + k0 + 8);
            // B lane data from global: K = kg + half*16 + {0..15}
            const int kg = kg0 + k0;
            v8h b0_lo = *(const v8h*)(wrow0 + kg);
            v8h b0_hi = *(const v8h*)(wrow0 + kg + 8);
            v8h b1_lo = *(const v8h*)(wrow1 + kg);
            v8h b1_hi = *(const v8h*)(wrow1 + kg + 8);
            v16h a, b0, b1;
#pragma unroll
            for (int i = 0; i < 8; ++i) {
                a[i]  = a_lo[i];  a[8 + i]  = a_hi[i];
                b0[i] = b0_lo[i]; b0[8 + i] = b0_hi[i];
                b1[i] = b1_lo[i]; b1[8 + i] = b1_hi[i];
            }
            c0 = __builtin_amdgcn_wmma_f32_16x16x32_f16(false, a, false, b0,
                                                        (short)0, c0, false, false);
            c1 = __builtin_amdgcn_wmma_f32_16x16x32_f16(false, a, false, b1,
                                                        (short)0, c1, false, false);
        }

        wait_fill();                             // prefetched LDS writes complete
        __syncthreads();                         // before anyone overwrites / reads
        buf ^= 1;
    }

    const int col0 = n0 + rc, col1 = col0 + 16;
#pragma unroll
    for (int r = 0; r < 8; ++r) {
        const int row = m0 + r + half * 8;       // C/D layout: lanes 16-31 -> M+8
        out[row * ldo + col0] = c0[r];
        out[row * ldo + col1] = c1[r];
        if (out16) {
            out16[row * N + col0] = (f16)c0[r];
            out16[row * N + col1] = (f16)c1[r];
        }
    }
}

// ---------------- elementwise / helper kernels ----------------
__global__ void cvt_f16_kernel(const float* __restrict__ s, f16* __restrict__ d, int n)
{
    int i = blockIdx.x * blockDim.x + threadIdx.x;
    if (i < n) d[i] = (f16)s[i];
}

__global__ void zero_state_kernel(float* h0, float* h1, f16* h016, f16* h116)
{
    int i = blockIdx.x * blockDim.x + threadIdx.x;
    if (i < NB * NH) {
        h0[i] = 0.f; h1[i] = 0.f;
        h016[i] = (f16)0.f; h116[i] = (f16)0.f;
    }
}

__global__ void init_tok_kernel(int* tok)
{
    if (threadIdx.x < NB) tok[threadIdx.x] = 0;
}

// encoder layer-0 input projection: K = 2, pure VALU
__global__ void enc_xproj_kernel(const float* __restrict__ x,
                                 const float* __restrict__ Wih0,
                                 const float* __restrict__ bih0,
                                 float* __restrict__ gih, int t)
{
    int i = blockIdx.x * blockDim.x + threadIdx.x;   // [0, B*3H)
    if (i >= NB * NH3) return;
    int b = i / NH3, g = i - b * NH3;
    float x0 = x[b * 2 * NS + t];
    float x1 = x[b * 2 * NS + NS + t];
    gih[i] = bih0[g] + Wih0[g * 2 + 0] * x0 + Wih0[g * 2 + 1] * x1;
}

// GRU gate math (PyTorch order r,z,n); optionally writes encoder-output slice
__global__ void gru_update_kernel(const float* __restrict__ gih,
                                  const float* __restrict__ ghh,
                                  float* __restrict__ h, f16* __restrict__ h16,
                                  float* __restrict__ enc_out, int t)
{
    int i = blockIdx.x * blockDim.x + threadIdx.x;   // [0, B*H)
    if (i >= NB * NH) return;
    int b = i / NH, j = i - b * NH;
    float xr = gih[b * NH3 + j];
    float xz = gih[b * NH3 + NH + j];
    float xn = gih[b * NH3 + 2 * NH + j];
    float hr = ghh[b * NH3 + j];
    float hz = ghh[b * NH3 + NH + j];
    float hn = ghh[b * NH3 + 2 * NH + j];
    float r = 1.f / (1.f + expf(-(xr + hr)));
    float z = 1.f / (1.f + expf(-(xz + hz)));
    float n = tanhf(xn + r * hn);
    float hv = (1.f - z) * n + z * h[i];
    h[i] = hv;
    h16[i] = (f16)hv;
    if (enc_out) enc_out[b * NS * NH + t * NH + j] = hv;
}

// decoder cell: h = tanh(gi + gh); optionally mirrors into cat[:, :H]
__global__ void dec_cell_kernel(const float* __restrict__ gi,
                                const float* __restrict__ gh,
                                float* __restrict__ h, f16* __restrict__ h16,
                                f16* __restrict__ cat16)
{
    int i = blockIdx.x * blockDim.x + threadIdx.x;   // [0, B*H)
    if (i >= NB * NH) return;
    int b = i / NH, j = i - b * NH;
    float v = tanhf(gi[i] + gh[i]);
    h[i] = v;
    h16[i] = (f16)v;
    if (cat16) cat16[b * 2 * NH + j] = (f16)v;
}

__global__ void embed_gather_kernel(const float* __restrict__ emb,
                                    const int* __restrict__ tok,
                                    f16* __restrict__ inp16)
{
    int i = blockIdx.x * blockDim.x + threadIdx.x;   // [0, B*H)
    if (i >= NB * NH) return;
    int b = i / NH, j = i - b * NH;
    inp16[i] = (f16)emb[tok[b] * NH + j];
}

// scores[b,s] = dot(enc_outs[b,s,:], q[b,:]) ; one wave per (b,s)
__global__ __launch_bounds__(256)
void attn_scores_kernel(const float* __restrict__ enc,
                        const float* __restrict__ q,
                        float* __restrict__ scores)
{
    int gw   = (blockIdx.x * 256 + threadIdx.x) >> 5;  // [0, B*S)
    int lane = threadIdx.x & 31;
    int b = gw / NS, s = gw - b * NS;
    const float* e  = enc + (size_t)b * NS * NH + (size_t)s * NH;
    const float* qb = q + b * NH;
    float acc = 0.f;
    for (int j = lane; j < NH; j += 32) acc += e[j] * qb[j];
#pragma unroll
    for (int off = 16; off > 0; off >>= 1) acc += __shfl_xor(acc, off, 32);
    if (lane == 0) scores[gw] = acc;
}

// row softmax over S=512; writes workspace aw and the attn output region
__global__ __launch_bounds__(512)
void softmax_rows_kernel(const float* __restrict__ scores,
                         float* __restrict__ aw,
                         float* __restrict__ attn_out)
{
    __shared__ float red[512];
    int b = blockIdx.x, s = threadIdx.x;
    float v = scores[b * NS + s];
    red[s] = v; __syncthreads();
    for (int off = 256; off > 0; off >>= 1) {
        if (s < off) red[s] = fmaxf(red[s], red[s + off]);
        __syncthreads();
    }
    float m = red[0]; __syncthreads();
    float e = expf(v - m);
    red[s] = e; __syncthreads();
    for (int off = 256; off > 0; off >>= 1) {
        if (s < off) red[s] += red[s + off];
        __syncthreads();
    }
    float a = e / red[0];
    aw[b * NS + s] = a;
    attn_out[b * NS + s] = a;
}

// att[b,h] = sum_s aw[b,s] * enc[b,s,h]  -> cat[:, H:2H]
__global__ void attn_combine_kernel(const float* __restrict__ enc,
                                    const float* __restrict__ aw,
                                    f16* __restrict__ cat16)
{
    int i = blockIdx.x * blockDim.x + threadIdx.x;   // [0, B*H)
    if (i >= NB * NH) return;
    int b = i / NH, j = i - b * NH;
    const float* e = enc + (size_t)b * NS * NH + j;
    const float* a = aw + b * NS;
    float acc = 0.f;
    for (int s = 0; s < NS; ++s) acc += a[s] * e[(size_t)s * NH];
    cat16[b * 2 * NH + NH + j] = (f16)acc;
}

// greedy argmax over V (first-max wins, matching jnp.argmax)
__global__ __launch_bounds__(256)
void argmax_rows_kernel(const float* __restrict__ logits, int ld, int* __restrict__ tok)
{
    __shared__ float sv[256];
    __shared__ int   si[256];
    int b = blockIdx.x, tn = threadIdx.x;
    const float* row = logits + (size_t)b * ld;
    float best = -3.4e38f; int bi = 0;
    for (int j = tn; j < NV; j += 256) {
        float v = row[j];
        if (v > best) { best = v; bi = j; }
    }
    sv[tn] = best; si[tn] = bi; __syncthreads();
    for (int off = 128; off > 0; off >>= 1) {
        if (tn < off) {
            if (sv[tn + off] > sv[tn] ||
                (sv[tn + off] == sv[tn] && si[tn + off] < si[tn])) {
                sv[tn] = sv[tn + off]; si[tn] = si[tn + off];
            }
        }
        __syncthreads();
    }
    if (tn == 0) tok[b] = si[0];
}

__global__ void copy_hidden_kernel(const float* __restrict__ h0,
                                   const float* __restrict__ h1,
                                   float* __restrict__ out)
{
    int i = blockIdx.x * blockDim.x + threadIdx.x;
    if (i < NB * NH) { out[i] = h0[i]; out[NB * NH + i] = h1[i]; }
}

// ---------------------------------------------------------------------------
extern "C" void kernel_launch(void* const* d_in, const int* in_sizes, int n_in,
                              void* d_out, int out_size, void* d_ws, size_t ws_size,
                              hipStream_t stream)
{
    const float* x     = (const float*)d_in[0];
    const float* emb   = (const float*)d_in[1];
    const float* eWih0 = (const float*)d_in[2];
    const float* eWhh0 = (const float*)d_in[3];
    const float* ebih0 = (const float*)d_in[4];
    const float* ebhh0 = (const float*)d_in[5];
    const float* dWih0 = (const float*)d_in[6];
    const float* dWhh0 = (const float*)d_in[7];
    const float* dbih0 = (const float*)d_in[8];
    const float* dbhh0 = (const float*)d_in[9];
    const float* eWih1 = (const float*)d_in[10];
    const float* eWhh1 = (const float*)d_in[11];
    const float* ebih1 = (const float*)d_in[12];
    const float* ebhh1 = (const float*)d_in[13];
    const float* dWih1 = (const float*)d_in[14];
    const float* dWhh1 = (const float*)d_in[15];
    const float* dbih1 = (const float*)d_in[16];
    const float* dbhh1 = (const float*)d_in[17];
    const float* hqW   = (const float*)d_in[18];
    const float* hqb   = (const float*)d_in[19];
    const float* combW = (const float*)d_in[20];
    const float* combb = (const float*)d_in[21];
    const float* fcW   = (const float*)d_in[22];
    const float* fcb   = (const float*)d_in[23];

    float* out = (float*)d_out;
    const size_t hiddenOff = (size_t)NB * NT * NV;                 // 4,194,304
    const size_t attnOff   = hiddenOff + (size_t)2 * NB * NH;      // +131,072

    // ---- workspace carve-out (256B aligned) ----
    char* wp = (char*)d_ws;
    auto carve = [&](size_t bytes) -> char* {
        char* p = wp; wp += (bytes + 255) & ~(size_t)255; return p;
    };
    float* encOut   = (float*)carve((size_t)NB * NS * NH * 4);     // 134 MB, L2-resident
    f16*   Whh0_16  = (f16*)carve((size_t)NH3 * NH * 2);
    f16*   Wih1_16  = (f16*)carve((size_t)NH3 * NH * 2);
    f16*   Whh1_16  = (f16*)carve((size_t)NH3 * NH * 2);
    f16*   dWih0_16 = (f16*)carve((size_t)NH * NH * 2);
    f16*   dWhh0_16 = (f16*)carve((size_t)NH * NH * 2);
    f16*   dWih1_16 = (f16*)carve((size_t)NH * NH * 2);
    f16*   dWhh1_16 = (f16*)carve((size_t)NH * NH * 2);
    f16*   hqW_16   = (f16*)carve((size_t)NH * NH * 2);
    f16*   combW_16 = (f16*)carve((size_t)NH * 2 * NH * 2);
    f16*   fcW_16   = (f16*)carve((size_t)NV * NH * 2);
    float* gih      = (float*)carve((size_t)NB * NH3 * 4);
    float* ghh      = (float*)carve((size_t)NB * NH3 * 4);
    float* h0       = (float*)carve((size_t)NB * NH * 4);
    float* h1       = (float*)carve((size_t)NB * NH * 4);
    f16*   h0_16    = (f16*)carve((size_t)NB * NH * 2);
    f16*   h1_16    = (f16*)carve((size_t)NB * NH * 2);
    float* gi       = (float*)carve((size_t)NB * NH * 4);
    float* gh       = (float*)carve((size_t)NB * NH * 4);
    float* qbuf     = (float*)carve((size_t)NB * NH * 4);
    f16*   inp16    = (f16*)carve((size_t)NB * NH * 2);
    f16*   cat16    = (f16*)carve((size_t)NB * 2 * NH * 2);
    float* c1       = (float*)carve((size_t)NB * NH * 4);
    f16*   c1_16    = (f16*)carve((size_t)NB * NH * 2);
    float* scores   = (float*)carve((size_t)NB * NS * 4);
    float* aw       = (float*)carve((size_t)NB * NS * 4);
    int*   tok      = (int*)carve(256);

    auto cvt = [&](const float* s, f16* d, int n) {
        cvt_f16_kernel<<<(n + 255) / 256, 256, 0, stream>>>(s, d, n);
    };
    auto gemm = [&](const f16* A, const f16* W, const float* bias,
                    float* o, int ldo, f16* o16, int N, int K) {
        gemm_wmma_f16<<<N / 64, 256, 0, stream>>>(A, W, bias, o, ldo, o16, N, K);
    };

    // ---- one-time weight conversion + state init ----
    cvt(eWhh0, Whh0_16, NH3 * NH);
    cvt(eWih1, Wih1_16, NH3 * NH);
    cvt(eWhh1, Whh1_16, NH3 * NH);
    cvt(dWih0, dWih0_16, NH * NH);
    cvt(dWhh0, dWhh0_16, NH * NH);
    cvt(dWih1, dWih1_16, NH * NH);
    cvt(dWhh1, dWhh1_16, NH * NH);
    cvt(hqW,   hqW_16,   NH * NH);
    cvt(combW, combW_16, NH * 2 * NH);
    cvt(fcW,   fcW_16,   NV * NH);
    zero_state_kernel<<<(NB * NH + 255) / 256, 256, 0, stream>>>(h0, h1, h0_16, h1_16);
    init_tok_kernel<<<1, 64, 0, stream>>>(tok);

    // ---- encoder: 512 sequential steps, 2 GRU layers ----
    for (int t = 0; t < NS; ++t) {
        enc_xproj_kernel<<<(NB * NH3) / 256, 256, 0, stream>>>(x, eWih0, ebih0, gih, t);
        gemm(h0_16, Whh0_16, ebhh0, ghh, NH3, nullptr, NH3, NH);
        gru_update_kernel<<<(NB * NH) / 256, 256, 0, stream>>>(gih, ghh, h0, h0_16, nullptr, t);
        gemm(h0_16, Wih1_16, ebih1, gih, NH3, nullptr, NH3, NH);
        gemm(h1_16, Whh1_16, ebhh1, ghh, NH3, nullptr, NH3, NH);
        gru_update_kernel<<<(NB * NH) / 256, 256, 0, stream>>>(gih, ghh, h1, h1_16, encOut, t);
    }

    // ---- decoder: 64 greedy autoregressive steps with attention ----
    for (int t = 0; t < NT; ++t) {
        embed_gather_kernel<<<(NB * NH) / 256, 256, 0, stream>>>(emb, tok, inp16);
        // layer 0
        gemm(inp16, dWih0_16, dbih0, gi, NH, nullptr, NH, NH);
        gemm(h0_16, dWhh0_16, dbhh0, gh, NH, nullptr, NH, NH);
        dec_cell_kernel<<<(NB * NH) / 256, 256, 0, stream>>>(gi, gh, h0, h0_16, nullptr);
        // layer 1 (output mirrored into cat[:, :H])
        gemm(h0_16, dWih1_16, dbih1, gi, NH, nullptr, NH, NH);
        gemm(h1_16, dWhh1_16, dbhh1, gh, NH, nullptr, NH, NH);
        dec_cell_kernel<<<(NB * NH) / 256, 256, 0, stream>>>(gi, gh, h1, h1_16, cat16);
        // attention
        gemm(h1_16, hqW_16, hqb, qbuf, NH, nullptr, NH, NH);
        attn_scores_kernel<<<(NB * NS) / 8, 256, 0, stream>>>(encOut, qbuf, scores);
        softmax_rows_kernel<<<NB, 512, 0, stream>>>(scores, aw, out + attnOff);
        attn_combine_kernel<<<(NB * NH) / 256, 256, 0, stream>>>(encOut, aw, cat16);
        // comb (K = 2H) then fc straight into d_out[:, t, :]
        gemm(cat16, combW_16, combb, c1, NH, c1_16, NH, 2 * NH);
        gemm(c1_16, fcW_16, fcb, out + (size_t)t * NV, NT * NV, nullptr, NV, NH);
        argmax_rows_kernel<<<NB, 256, 0, stream>>>(out + (size_t)t * NV, NT * NV, tok);
    }

    copy_hidden_kernel<<<(NB * NH) / 256, 256, 0, stream>>>(h0, h1, out + hiddenOff);
}
